// DilatedAttention_61890478735485
// MI455X (gfx1250) — compile-verified, matrix operand order checked by reading, not run on hardware
//
#include <hip/hip_runtime.h>
#include <hip/hip_bf16.h>
#include <math.h>

// ---------------------------------------------------------------------------
// Dilated attention on gfx1250 (MI455X): f16 WMMA everywhere, fp32 accumulate.
// GEMMs use double-buffered GLOBAL_LOAD_ASYNC_TO_LDS (ASYNCcnt) when available.
// ---------------------------------------------------------------------------

typedef __attribute__((ext_vector_type(16))) _Float16 v16h;
typedef __attribute__((ext_vector_type(8)))  _Float16 v8h;
typedef __attribute__((ext_vector_type(4)))  _Float16 v4h;
typedef __attribute__((ext_vector_type(8)))  float    v8f;
typedef __attribute__((ext_vector_type(4)))  int      v4i;

typedef __attribute__((address_space(1))) v4i gas_v4i;   // global
typedef __attribute__((address_space(3))) v4i las_v4i;   // LDS

#define DM      1024          // d_model
#define D3      3072          // 3 * d_model
#define HEADS16 16            // heads per (b,n)
#define HD      64            // head dim
#define LQ      1024          // dilated segment length (SEG / DILATION)
#define MROWS   8192          // B * N * LQ active rows
#define NHEADS  128           // B * N * HEADS16

// Workspace byte offsets (256-aligned)
#define OFF_XG     ((size_t)0)            // [8192][1024] f16
#define OFF_WQKVT  ((size_t)16777216)     // [3072][1024] f16 (transposed)
#define OFF_WOT    ((size_t)23068672)     // [1024][1024] f16 (transposed)
#define OFF_Q      ((size_t)25165824)     // [128][1024][64] f16
#define OFF_K      ((size_t)41943040)     // [128][1024][64] f16
#define OFF_V      ((size_t)58720256)     // [128][64][1024] f16 (transposed)
#define OFF_CTX    ((size_t)75497472)     // [8192][1024] f16

// ---------------------------------------------------------------------------
// Async global -> LDS copy (CDNA5 ASYNCcnt path) with sync fallback
// ---------------------------------------------------------------------------
#if __has_builtin(__builtin_amdgcn_global_load_async_to_lds_b128)
#define HAVE_ASYNC_LDS 1
#else
#define HAVE_ASYNC_LDS 0
#endif

#if HAVE_ASYNC_LDS
#if __has_builtin(__builtin_amdgcn_s_wait_asynccnt)
#define WAIT_ASYNC(n) __builtin_amdgcn_s_wait_asynccnt(n)
#else
#define WAIT_ASYNC(n) asm volatile("s_wait_asynccnt %0" ::"n"(n))
#endif
#else
#define WAIT_ASYNC(n)
#endif

__device__ __forceinline__ void async_copy16(const _Float16* g, _Float16* l) {
#if HAVE_ASYNC_LDS
  __builtin_amdgcn_global_load_async_to_lds_b128(
      (gas_v4i*)g, (las_v4i*)l, 0, 0);
#else
  *(v8h*)l = *(const v8h*)g;
#endif
}

// ---------------------------------------------------------------------------
// WMMA helpers
// ---------------------------------------------------------------------------
__device__ __forceinline__ v8f wmma16(v16h a, v16h b, v8f c) {
  return __builtin_amdgcn_wmma_f32_16x16x32_f16(
      /*neg_a=*/false, a, /*neg_b=*/false, b,
      /*c_mod=*/(short)0, c, /*reuse_a=*/false, /*reuse_b=*/false);
}

// Load a 16x32 f16 A-style fragment from a row-major matrix.
// `base` points at (row 0, k 0) of the fragment; `stride` in halves.
// ISA layout: lane L holds row (L&15); lanes 0-15 carry K 0..7 / 16..23,
// lanes 16-31 carry K 8..15 / 24..31  (B fragments == A fragment of B^T).
__device__ __forceinline__ v16h load_frag(const _Float16* base, int stride) {
  int lane = threadIdx.x & 31;
  int r    = lane & 15;
  int ko   = (lane >> 4) << 3;            // 0 or 8
  const _Float16* p = base + (size_t)r * stride + ko;
  v8h lo = *(const v8h*)(p);
  v8h hi = *(const v8h*)(p + 16);
  v16h f;
#pragma unroll
  for (int i = 0; i < 8; ++i) { f[i] = lo[i]; f[i + 8] = hi[i]; }
  return f;
}

__device__ __forceinline__ float hmax16(float x) {
  x = fmaxf(x, __shfl_xor(x, 1));
  x = fmaxf(x, __shfl_xor(x, 2));
  x = fmaxf(x, __shfl_xor(x, 4));
  x = fmaxf(x, __shfl_xor(x, 8));
  return x;
}
__device__ __forceinline__ float hsum16(float x) {
  x += __shfl_xor(x, 1);
  x += __shfl_xor(x, 2);
  x += __shfl_xor(x, 4);
  x += __shfl_xor(x, 8);
  return x;
}

// ---------------------------------------------------------------------------
// Prep: gather dilated rows of x -> f16   (row r = (b, n, l), src row l*2)
// ---------------------------------------------------------------------------
__global__ __launch_bounds__(256)
void k_gather_x(const float* __restrict__ x, _Float16* __restrict__ xg) {
  int r   = blockIdx.x;                  // 0..8191
  int b   = r >> 12;
  int rem = r & 4095;
  int n   = rem >> 10;
  int l   = rem & 1023;
  const float* src = x + ((size_t)(b * 8192 + n * 2048 + l * 2) << 10);
  _Float16*    dst = xg + ((size_t)r << 10);
  int c = threadIdx.x * 4;
  float4 f = *(const float4*)(src + c);
  v4h h; h[0] = (_Float16)f.x; h[1] = (_Float16)f.y;
         h[2] = (_Float16)f.z; h[3] = (_Float16)f.w;
  *(v4h*)(dst + c) = h;
}

// Transpose-convert weight: src fp32 [K][N] -> dst f16 [N][K]
__global__ __launch_bounds__(256)
void k_transpose_w(const float* __restrict__ src, _Float16* __restrict__ dst,
                   int K, int N) {
  int c = blockIdx.x;                    // column of src = row of dst
  for (int k = threadIdx.x; k < K; k += 256)
    dst[(size_t)c * K + k] = (_Float16)src[(size_t)k * N + c];
}

// ---------------------------------------------------------------------------
// Stage 1: QKV GEMM  C[8192][3072] = xg @ wqkv + bqkv, scattered to Q/K/V^T
// 64x64 workgroup tile, 4 waves (32x32 per wave), double-buffered async LDS.
// ---------------------------------------------------------------------------
__global__ __launch_bounds__(128)
void k_qkv_gemm(const _Float16* __restrict__ A,      // xg   [8192][1024]
                const _Float16* __restrict__ BT,     // wqkvT[3072][1024]
                const float*    __restrict__ bias,   // bqkv [3072]
                _Float16* __restrict__ qws,
                _Float16* __restrict__ kws,
                _Float16* __restrict__ vws) {
  __shared__ _Float16 As[2][64][40];
  __shared__ _Float16 Bs[2][64][40];

  int m0 = blockIdx.x * 64;
  int n0 = blockIdx.y * 64;
  int t  = threadIdx.x;
  int w  = t >> 5;
  int sm = (w >> 1) * 32, sn = (w & 1) * 32;
  int srow = t >> 1, scol = (t & 1) * 16;

  v8f acc[4];
#pragma unroll
  for (int i = 0; i < 4; ++i) acc[i] = (v8f)(0.0f);

  auto issue = [&](int buf, int kc) {
    const _Float16* ga = A  + (size_t)(m0 + srow) * DM + kc + scol;
    const _Float16* gb = BT + (size_t)(n0 + srow) * DM + kc + scol;
    async_copy16(ga,     &As[buf][srow][scol]);
    async_copy16(ga + 8, &As[buf][srow][scol + 8]);
    async_copy16(gb,     &Bs[buf][srow][scol]);
    async_copy16(gb + 8, &Bs[buf][srow][scol + 8]);
  };

  issue(0, 0);
  int cur = 0;
  for (int kc = 0; kc < DM; kc += 32) {
    if (kc + 32 < DM) { issue(cur ^ 1, kc + 32); WAIT_ASYNC(4); }
    else              { WAIT_ASYNC(0); }
    __syncthreads();

    v16h af0 = load_frag(&As[cur][sm][0],      40);
    v16h af1 = load_frag(&As[cur][sm + 16][0], 40);
    v16h bf0 = load_frag(&Bs[cur][sn][0],      40);
    v16h bf1 = load_frag(&Bs[cur][sn + 16][0], 40);
    acc[0] = wmma16(af0, bf0, acc[0]);
    acc[1] = wmma16(af0, bf1, acc[1]);
    acc[2] = wmma16(af1, bf0, acc[2]);
    acc[3] = wmma16(af1, bf1, acc[3]);
    __syncthreads();
    cur ^= 1;
  }

  // Epilogue: add bias, scatter to per-head Q/K and transposed V
  int lane = t & 31;
  int ncol = lane & 15;
  int hi8  = (lane >> 4) << 3;
#pragma unroll
  for (int ti = 0; ti < 2; ++ti) {
#pragma unroll
    for (int tj = 0; tj < 2; ++tj) {
      v8f cacc = acc[ti * 2 + tj];
      int c_g = n0 + sn + tj * 16 + ncol;
      float bv = bias[c_g];
      int which = c_g >> 10;           // 0=q 1=k 2=v
      int cc = c_g & 1023;
      int h  = cc >> 6;
      int d  = cc & 63;
#pragma unroll
      for (int v = 0; v < 8; ++v) {
        int m_g = m0 + sm + ti * 16 + v + hi8;
        int bn  = m_g >> 10;
        int l   = m_g & 1023;
        int head = bn * HEADS16 + h;
        _Float16 hv = (_Float16)(cacc[v] + bv);
        if (which == 0)       qws[((size_t)head << 16) + (size_t)l * HD + d] = hv;
        else if (which == 1)  kws[((size_t)head << 16) + (size_t)l * HD + d] = hv;
        else                  vws[((size_t)head << 16) + (size_t)d * LQ + l] = hv;
      }
    }
  }
}

// ---------------------------------------------------------------------------
// Stage 2: flash attention.  One wave per (head, 16-query block).
// grid = 128 heads * 64 qblocks = 8192 blocks of 32 threads.
// ---------------------------------------------------------------------------
__global__ __launch_bounds__(32)
void k_flash(const _Float16* __restrict__ qws,
             const _Float16* __restrict__ kws,
             const _Float16* __restrict__ vws,   // transposed [head][64][1024]
             _Float16* __restrict__ ctx) {       // [8192][1024]
  __shared__ _Float16 plds[16][40];

  int head = blockIdx.x >> 6;
  int qb   = blockIdx.x & 63;
  int lane = threadIdx.x & 31;
  int ncol = lane & 15;
  int hi8  = (lane >> 4) << 3;

  const _Float16* qbase = qws + ((size_t)head << 16) + (size_t)qb * 16 * HD;
  const _Float16* kbase = kws + ((size_t)head << 16);
  const _Float16* vbase = vws + ((size_t)head << 16);

  // Q fragments (constant across the key loop), pre-scaled by 1/sqrt(hd)
  v16h aq0 = load_frag(qbase, HD);
  v16h aq1 = load_frag(qbase + 32, HD);
  const _Float16 sc = (_Float16)0.125f;
#pragma unroll
  for (int i = 0; i < 16; ++i) { aq0[i] *= sc; aq1[i] *= sc; }

  float rowmax[8], rowsum[8];
  v8f acc[4];
#pragma unroll
  for (int v = 0; v < 8; ++v) { rowmax[v] = -__builtin_inff(); rowsum[v] = 0.0f; }
#pragma unroll
  for (int j = 0; j < 4; ++j) acc[j] = (v8f)(0.0f);

  for (int kc = 0; kc < LQ; kc += 32) {
    // S = (Q/sqrt(hd)) @ K^T for 32 keys -> two 16x16 f32 tiles
    v8f s0 = (v8f)(0.0f), s1 = (v8f)(0.0f);
    s0 = wmma16(aq0, load_frag(kbase + (size_t)(kc)      * HD,      HD), s0);
    s0 = wmma16(aq1, load_frag(kbase + (size_t)(kc)      * HD + 32, HD), s0);
    s1 = wmma16(aq0, load_frag(kbase + (size_t)(kc + 16) * HD,      HD), s1);
    s1 = wmma16(aq1, load_frag(kbase + (size_t)(kc + 16) * HD + 32, HD), s1);

    // Online softmax stats (per output row m = v + hi8)
    float al[8];
#pragma unroll
    for (int v = 0; v < 8; ++v) {
      float mx = hmax16(fmaxf(s0[v], s1[v]));
      float nm = fmaxf(rowmax[v], mx);
      float a  = __expf(rowmax[v] - nm);
      rowmax[v] = nm;
      float p0 = __expf(s0[v] - nm);
      float p1 = __expf(s1[v] - nm);
      s0[v] = p0; s1[v] = p1;
      rowsum[v] = rowsum[v] * a + hsum16(p0 + p1);
      al[v] = a;
    }
#pragma unroll
    for (int j = 0; j < 4; ++j)
#pragma unroll
      for (int v = 0; v < 8; ++v) acc[j][v] *= al[v];

    // C-layout P -> A-layout fragment via LDS round-trip
#pragma unroll
    for (int v = 0; v < 8; ++v) {
      int m = v + hi8;
      plds[m][ncol]      = (_Float16)s0[v];
      plds[m][ncol + 16] = (_Float16)s1[v];
    }
    __syncthreads();
    v16h ap = load_frag(&plds[0][0], 40);

    // acc += P @ V  (B^T = V^T stored row-major [d][key])
#pragma unroll
    for (int j = 0; j < 4; ++j) {
      v16h bv = load_frag(vbase + (size_t)(j * 16) * LQ + kc, LQ);
      acc[j] = wmma16(ap, bv, acc[j]);
    }
    __syncthreads();
  }

  // Normalize and write context (f16, row-major [8192][1024])
  int bn = head >> 4, h = head & 15;
  size_t rowbase = (size_t)(bn * LQ + qb * 16);
#pragma unroll
  for (int v = 0; v < 8; ++v) {
    float inv = 1.0f / rowsum[v];
    int m = v + hi8;
    size_t row = (rowbase + m) * DM + h * HD;
#pragma unroll
    for (int j = 0; j < 4; ++j)
      ctx[row + j * 16 + ncol] = (_Float16)(acc[j][v] * inv);
  }
}

// ---------------------------------------------------------------------------
// Stage 3: output GEMM  out[8192][1024] = ctx @ wo + bo   (fp32 out)
// ---------------------------------------------------------------------------
__global__ __launch_bounds__(128)
void k_out_gemm(const _Float16* __restrict__ A,     // ctx  [8192][1024]
                const _Float16* __restrict__ BT,    // woT  [1024][1024]
                const float*    __restrict__ bias,  // bo   [1024]
                float* __restrict__ out) {
  __shared__ _Float16 As[2][64][40];
  __shared__ _Float16 Bs[2][64][40];

  int m0 = blockIdx.x * 64;
  int n0 = blockIdx.y * 64;
  int t  = threadIdx.x;
  int w  = t >> 5;
  int sm = (w >> 1) * 32, sn = (w & 1) * 32;
  int srow = t >> 1, scol = (t & 1) * 16;

  v8f acc[4];
#pragma unroll
  for (int i = 0; i < 4; ++i) acc[i] = (v8f)(0.0f);

  auto issue = [&](int buf, int kc) {
    const _Float16* ga = A  + (size_t)(m0 + srow) * DM + kc + scol;
    const _Float16* gb = BT + (size_t)(n0 + srow) * DM + kc + scol;
    async_copy16(ga,     &As[buf][srow][scol]);
    async_copy16(ga + 8, &As[buf][srow][scol + 8]);
    async_copy16(gb,     &Bs[buf][srow][scol]);
    async_copy16(gb + 8, &Bs[buf][srow][scol + 8]);
  };

  issue(0, 0);
  int cur = 0;
  for (int kc = 0; kc < DM; kc += 32) {
    if (kc + 32 < DM) { issue(cur ^ 1, kc + 32); WAIT_ASYNC(4); }
    else              { WAIT_ASYNC(0); }
    __syncthreads();

    v16h af0 = load_frag(&As[cur][sm][0],      40);
    v16h af1 = load_frag(&As[cur][sm + 16][0], 40);
    v16h bf0 = load_frag(&Bs[cur][sn][0],      40);
    v16h bf1 = load_frag(&Bs[cur][sn + 16][0], 40);
    acc[0] = wmma16(af0, bf0, acc[0]);
    acc[1] = wmma16(af0, bf1, acc[1]);
    acc[2] = wmma16(af1, bf0, acc[2]);
    acc[3] = wmma16(af1, bf1, acc[3]);
    __syncthreads();
    cur ^= 1;
  }

  int lane = t & 31;
  int ncol = lane & 15;
  int hi8  = (lane >> 4) << 3;
#pragma unroll
  for (int ti = 0; ti < 2; ++ti) {
#pragma unroll
    for (int tj = 0; tj < 2; ++tj) {
      v8f cacc = acc[ti * 2 + tj];
      int c_g = n0 + sn + tj * 16 + ncol;
      float bv = bias[c_g];
#pragma unroll
      for (int v = 0; v < 8; ++v) {
        int m_g = m0 + sm + ti * 16 + v + hi8;
        out[(size_t)m_g * DM + c_g] = cacc[v] + bv;
      }
    }
  }
}

// ---------------------------------------------------------------------------
// Host launcher
// ---------------------------------------------------------------------------
extern "C" void kernel_launch(void* const* d_in, const int* in_sizes, int n_in,
                              void* d_out, int out_size, void* d_ws, size_t ws_size,
                              hipStream_t stream) {
  const float* x    = (const float*)d_in[0];
  const float* wqkv = (const float*)d_in[1];
  const float* bqkv = (const float*)d_in[2];
  const float* wo   = (const float*)d_in[3];
  const float* bo   = (const float*)d_in[4];
  float* out = (float*)d_out;

  char* ws = (char*)d_ws;
  _Float16* xg     = (_Float16*)(ws + OFF_XG);
  _Float16* wqkvT  = (_Float16*)(ws + OFF_WQKVT);
  _Float16* woT    = (_Float16*)(ws + OFF_WOT);
  _Float16* qws    = (_Float16*)(ws + OFF_Q);
  _Float16* kws    = (_Float16*)(ws + OFF_K);
  _Float16* vws    = (_Float16*)(ws + OFF_V);
  _Float16* ctx    = (_Float16*)(ws + OFF_CTX);

  // Prep
  k_gather_x<<<MROWS, 256, 0, stream>>>(x, xg);
  k_transpose_w<<<D3, 256, 0, stream>>>(wqkv, wqkvT, DM, D3);
  k_transpose_w<<<DM, 256, 0, stream>>>(wo, woT, DM, DM);

  // Stage 1: QKV projection
  k_qkv_gemm<<<dim3(MROWS / 64, D3 / 64), 128, 0, stream>>>(
      xg, wqkvT, bqkv, qws, kws, vws);

  // Stage 2: flash attention (128 heads x 64 query blocks)
  k_flash<<<NHEADS * 64, 32, 0, stream>>>(qws, kws, vws, ctx);

  // Stage 3: output projection
  k_out_gemm<<<dim3(MROWS / 64, DM / 64), 128, 0, stream>>>(ctx, woT, bo, out);
}
